// BasicCNN_22479858827954
// MI455X (gfx1250) — compile-verified, matrix operand order checked by reading, not run on hardware
//
#include <hip/hip_runtime.h>
#include <hip/hip_bf16.h>

// ---------- types ----------
typedef __attribute__((ext_vector_type(16))) __bf16 bf16x16;
typedef __attribute__((ext_vector_type(8)))  __bf16 bf16x8;
typedef __attribute__((ext_vector_type(8)))  float  f32x8;

#define TOT   4096
#define SEN   1024
#define OUTD  1024
#define LORA_R 64
#define LORA_SCALE 2.0f        // 128/64
#define CNN_OUT 3264
#define NUM_OUT 1968
#define BATCH 1024

// =====================================================================
// Kernel 1: W_eff^T (bf16) = ((W + lora_A @ lora_B * scale) * mask)^T
// =====================================================================
__global__ void build_weff(const float* __restrict__ W,
                           const float* __restrict__ lA,
                           const float* __restrict__ lB,
                           __bf16* __restrict__ WT) {
  __shared__ float sA[32][64];
  __shared__ float sB[64][33];
  const int k0 = blockIdx.y * 32, n0 = blockIdx.x * 32;
  const int tid = threadIdx.x;      // 256 threads
  for (int i = tid; i < 32 * 64; i += 256) {
    int r = i >> 6, c = i & 63;
    sA[r][c] = lA[(size_t)(k0 + r) * LORA_R + c];
  }
  for (int i = tid; i < 64 * 32; i += 256) {
    int r = i >> 5, c = i & 31;
    sB[r][c] = lB[(size_t)r * TOT + n0 + c];
  }
  __syncthreads();
  for (int i = tid; i < 1024; i += 256) {
    int kk = i >> 5, nn = i & 31;
    int k = k0 + kk, n = n0 + nn;
    float w = W[(size_t)k * TOT + n];
    float v = 0.f;
    if (w != 0.f) {
      float acc = 0.f;
      #pragma unroll
      for (int r = 0; r < LORA_R; ++r) acc = fmaf(sA[kk][r], sB[r][nn], acc);
      v = w + LORA_SCALE * acc;
    }
    WT[(size_t)n * TOT + k] = (__bf16)v;
  }
}

// =====================================================================
// Kernel 2: transpose f32 [K][N] -> bf16 [N][K]
// =====================================================================
__global__ void transpose_cvt(const float* __restrict__ in,
                              __bf16* __restrict__ out, int K, int N) {
  __shared__ float tile[32][33];
  const int kb = blockIdx.y * 32, nb = blockIdx.x * 32;
  const int tx = threadIdx.x & 31, ty = threadIdx.x >> 5;
  for (int i = ty; i < 32; i += 8) {
    int k = kb + i, n = nb + tx;
    tile[i][tx] = (k < K && n < N) ? in[(size_t)k * N + n] : 0.f;
  }
  __syncthreads();
  for (int i = ty; i < 32; i += 8) {
    int n = nb + i, k = kb + tx;
    if (n < N && k < K) out[(size_t)n * K + k] = (__bf16)tile[tx][i];
  }
}

// =====================================================================
// Kernel 3: multi-scale valid conv + ReLU -> feat bf16 [1024][3264]
// =====================================================================
__global__ void conv_feat(const float* __restrict__ x,
                          const float* __restrict__ w1, const float* __restrict__ w2,
                          const float* __restrict__ w3, const float* __restrict__ w4,
                          const float* __restrict__ w5, const float* __restrict__ w6,
                          const float* __restrict__ w7, const float* __restrict__ w8,
                          const float* __restrict__ cb,
                          __bf16* __restrict__ feat) {
  __shared__ float xs[8 * 8 * 8];
  const int b = blockIdx.x, tid = threadIdx.x;   // 256 threads
  for (int i = tid; i < 512; i += 256) xs[i] = x[(size_t)b * 512 + i];
  __syncthreads();
  const float* wk[8] = {w1, w2, w3, w4, w5, w6, w7, w8};
  const int base[9] = {0, 1024, 1808, 2384, 2784, 3040, 3184, 3248, 3264};
  for (int fi = tid; fi < CNN_OUT; fi += 256) {
    int k = 1;
    while (fi >= base[k]) ++k;
    const int local = fi - base[k - 1];
    const int o = 9 - k;
    const int f = local / (o * o);
    const int rem = local - f * o * o;
    const int oi = rem / o, oj = rem - oi * o;
    const float* w = wk[k - 1];
    float acc = cb[(k - 1) * 16 + f];
    for (int ki = 0; ki < k; ++ki)
      for (int kj = 0; kj < k; ++kj) {
        const float* xp = &xs[(((oi + ki) * 8) + (oj + kj)) * 8];
        const float* wp = &w[((f * 8) * k + ki) * k + kj];
        #pragma unroll
        for (int c = 0; c < 8; ++c) acc = fmaf(xp[c], wp[(size_t)c * k * k], acc);
      }
    acc = acc > 0.f ? acc : 0.f;
    feat[(size_t)b * CNN_OUT + fi] = (__bf16)acc;
  }
}

// =====================================================================
// Kernel 4: state init: state = relu(pad(E)); also bf16 mirror
// =====================================================================
__global__ void init_state(const float* __restrict__ E,
                           float* __restrict__ s, __bf16* __restrict__ sb) {
  const size_t idx = (size_t)blockIdx.x * 256 + threadIdx.x;
  const int c = (int)(idx & (TOT - 1));
  const int b = (int)(idx >> 12);
  float v = 0.f;
  if (c < SEN) {
    v = E[(size_t)b * SEN + c];
    v = v > 0.f ? v : 0.f;
  }
  s[idx] = v;
  sb[idx] = (__bf16)v;
}

// =====================================================================
// Kernel 5: bf16 WMMA GEMM, 128(M)x256(N) tile, 8 waves (2x4),
//   64x64 per wave -> 16 v_wmma per wave per K-step(32).
//   UNBOUND path: async global->LDS DMA (ASYNCcnt) + double buffering.
//   BOUND path (ragged N): register-staged, guarded, zero-filled.
// =====================================================================
#define TM 128
#define TN 256
#define GSTR 40                     // LDS row stride in bf16 elems
#define BUF_E ((TM + TN) * GSTR)    // elems per LDS buffer (A then B)
#define BUF_B (BUF_E * 2)           // bytes per LDS buffer

__device__ __forceinline__ bf16x16 load_frag(const __bf16* p, int half) {
  // lanes 0-15 hold K {0..7,16..23}; lanes 16-31 hold K {8..15,24..31}
  union { bf16x16 v; bf16x8 h[2]; } u;
  u.h[0] = *(const bf16x8*)(p + half * 8);
  u.h[1] = *(const bf16x8*)(p + 16 + half * 8);
  return u.v;
}

__device__ __forceinline__ void async_ld16(unsigned lds_byte, const void* g) {
  // CDNA5 async DMA: 16B global -> LDS, tracked by ASYNCcnt
  asm volatile("global_load_async_to_lds_b128 %0, %1, off"
               :: "v"(lds_byte), "v"(g) : "memory");
}
__device__ __forceinline__ void wait_async0() {
  asm volatile("s_wait_asynccnt 0x0" ::: "memory");
}

template <bool BOUND>
__global__ void __launch_bounds__(256)
gemm_bf16(const __bf16* __restrict__ A, int lda,
          const __bf16* __restrict__ Bt, int ldb,
          const float* __restrict__ bias,
          const float* __restrict__ residual, int ldr,
          float* __restrict__ Cf, int ldc,
          __bf16* __restrict__ Cb, int ldcb,
          int M, int N, int K, int relu) {
  __shared__ __bf16 smem[2 * BUF_E];   // double buffer: [A 128 rows | B 256 rows]
  const int tid = threadIdx.x;
  const int wave = tid >> 5, lane = tid & 31;
  const int half = lane >> 4, l15 = lane & 15;
  const int waveM = wave >> 2;          // 0..1 -> 64 rows
  const int waveN = wave & 3;           // 0..3 -> 64 cols
  const int m0 = blockIdx.y * TM, n0 = blockIdx.x * TN;

  // staging: thread owns rows (sr, sr+64) of A and (sr,+64,+128,+192) of Bt
  const int sr = tid >> 2;              // 0..63
  const int kc = (tid & 3) * 8;         // K sub-chunk (8 bf16 = 16 B)
  const __bf16* aP0 = A  + (size_t)(m0 + sr      ) * lda + kc;
  const __bf16* aP1 = A  + (size_t)(m0 + sr + 64 ) * lda + kc;
  const __bf16* bP0 = Bt + (size_t)(n0 + sr      ) * ldb + kc;
  const __bf16* bP1 = Bt + (size_t)(n0 + sr + 64 ) * ldb + kc;
  const __bf16* bP2 = Bt + (size_t)(n0 + sr + 128) * ldb + kc;
  const __bf16* bP3 = Bt + (size_t)(n0 + sr + 192) * ldb + kc;
  // LDS byte offsets (buffer 0) for this thread's six 16B chunks
  const unsigned oA0 = (unsigned)((sr      ) * GSTR + kc) * 2u;
  const unsigned oA1 = (unsigned)((sr + 64 ) * GSTR + kc) * 2u;
  const unsigned oB0 = (unsigned)((TM + sr      ) * GSTR + kc) * 2u;
  const unsigned oB1 = (unsigned)((TM + sr + 64 ) * GSTR + kc) * 2u;
  const unsigned oB2 = (unsigned)((TM + sr + 128) * GSTR + kc) * 2u;
  const unsigned oB3 = (unsigned)((TM + sr + 192) * GSTR + kc) * 2u;

  f32x8 acc[4][4] = {};   // 64x64 per wave
  int cur = 0;

  if constexpr (!BOUND) {
    // ---------------- async-DMA double-buffered main loop ----------------
    // prologue: slice 0 -> buffer 0
    async_ld16(oA0, aP0); async_ld16(oA1, aP1);
    async_ld16(oB0, bP0); async_ld16(oB1, bP1);
    async_ld16(oB2, bP2); async_ld16(oB3, bP3);
    for (int kb = 0; kb < K; kb += 32) {
      wait_async0();          // this wave's chunks landed
      __syncthreads();        // every wave's chunks landed
      const int kn = kb + 32;
      if (kn < K) {           // refill the other buffer under the WMMAs
        const unsigned bo = (unsigned)(cur ^ 1) * (unsigned)BUF_B;
        async_ld16(bo + oA0, aP0 + kn); async_ld16(bo + oA1, aP1 + kn);
        async_ld16(bo + oB0, bP0 + kn); async_ld16(bo + oB1, bP1 + kn);
        async_ld16(bo + oB2, bP2 + kn); async_ld16(bo + oB3, bP3 + kn);
      }
      const __bf16* buf = smem + cur * BUF_E;
      bf16x16 bfr[4];
      #pragma unroll
      for (int n = 0; n < 4; ++n)
        bfr[n] = load_frag(buf + (TM + waveN * 64 + n * 16 + l15) * GSTR, half);
      #pragma unroll
      for (int m = 0; m < 4; ++m) {
        bf16x16 af = load_frag(buf + (waveM * 64 + m * 16 + l15) * GSTR, half);
        #pragma unroll
        for (int n = 0; n < 4; ++n)
          acc[m][n] = __builtin_amdgcn_wmma_f32_16x16x32_bf16(
              false, af, false, bfr[n], (short)0, acc[m][n], false, false);
      }
      cur ^= 1;
    }
  } else {
    // ---------------- guarded register-staged loop (ragged N) ----------------
    const bool bv0 = (n0 + sr       < N);
    const bool bv1 = (n0 + sr + 64  < N);
    const bool bv2 = (n0 + sr + 128 < N);
    const bool bv3 = (n0 + sr + 192 < N);
    const uint4 zz = {0u, 0u, 0u, 0u};
    uint4 ga0 = *(const uint4*)(aP0);
    uint4 ga1 = *(const uint4*)(aP1);
    uint4 gb0 = bv0 ? *(const uint4*)(bP0) : zz;
    uint4 gb1 = bv1 ? *(const uint4*)(bP1) : zz;
    uint4 gb2 = bv2 ? *(const uint4*)(bP2) : zz;
    uint4 gb3 = bv3 ? *(const uint4*)(bP3) : zz;
    for (int kb = 0; kb < K; kb += 32) {
      __bf16* buf = smem + cur * BUF_E;
      *(uint4*)((char*)buf + oA0) = ga0;
      *(uint4*)((char*)buf + oA1) = ga1;
      *(uint4*)((char*)buf + oB0) = gb0;
      *(uint4*)((char*)buf + oB1) = gb1;
      *(uint4*)((char*)buf + oB2) = gb2;
      *(uint4*)((char*)buf + oB3) = gb3;
      __syncthreads();
      const int kn = kb + 32;
      if (kn < K) {
        ga0 = *(const uint4*)(aP0 + kn);
        ga1 = *(const uint4*)(aP1 + kn);
        gb0 = bv0 ? *(const uint4*)(bP0 + kn) : zz;
        gb1 = bv1 ? *(const uint4*)(bP1 + kn) : zz;
        gb2 = bv2 ? *(const uint4*)(bP2 + kn) : zz;
        gb3 = bv3 ? *(const uint4*)(bP3 + kn) : zz;
      }
      bf16x16 bfr[4];
      #pragma unroll
      for (int n = 0; n < 4; ++n)
        bfr[n] = load_frag(buf + (TM + waveN * 64 + n * 16 + l15) * GSTR, half);
      #pragma unroll
      for (int m = 0; m < 4; ++m) {
        bf16x16 af = load_frag(buf + (waveM * 64 + m * 16 + l15) * GSTR, half);
        #pragma unroll
        for (int n = 0; n < 4; ++n)
          acc[m][n] = __builtin_amdgcn_wmma_f32_16x16x32_bf16(
              false, af, false, bfr[n], (short)0, acc[m][n], false, false);
      }
      __syncthreads();
      cur ^= 1;
    }
  }

  // epilogue; C layout: VGPR r -> M = r (lanes 0-15) / r+8 (lanes 16-31)
  #pragma unroll
  for (int m = 0; m < 4; ++m) {
    const int rl = waveM * 64 + m * 16 + half * 8;
    #pragma unroll
    for (int n = 0; n < 4; ++n) {
      const int col = n0 + waveN * 64 + n * 16 + l15;
      #pragma unroll
      for (int r = 0; r < 8; ++r) {
        const int row = m0 + rl + r;
        if (!BOUND || (row < M && col < N)) {
          float v = acc[m][n][r];
          if (bias)     v += bias[col];
          if (residual) v += residual[(size_t)row * ldr + col];
          if (relu)     v = v > 0.f ? v : 0.f;
          if (Cf) Cf[(size_t)row * ldc + col] = v;
          if (Cb) Cb[(size_t)row * ldcb + col] = (__bf16)v;
        }
      }
    }
  }
}

// =====================================================================
// Host launcher
// =====================================================================
extern "C" void kernel_launch(void* const* d_in, const int* in_sizes, int n_in,
                              void* d_out, int out_size, void* d_ws, size_t ws_size,
                              hipStream_t stream) {
  (void)in_sizes; (void)n_in; (void)out_size; (void)ws_size;
  const float* x      = (const float*)d_in[0];
  const float* cw[8];
  for (int i = 0; i < 8; ++i) cw[i] = (const float*)d_in[1 + i];
  const float* conv_b = (const float*)d_in[9];
  const float* W      = (const float*)d_in[10];
  const float* lora_A = (const float*)d_in[11];
  const float* lora_B = (const float*)d_in[12];
  const float* ip_w   = (const float*)d_in[13];
  const float* ip_b   = (const float*)d_in[14];
  const float* out_w  = (const float*)d_in[15];
  const float* out_b  = (const float*)d_in[16];
  float* out = (float*)d_out;

  size_t off = 0;
  auto carve = [&](size_t bytes) -> void* {
    void* p = (char*)d_ws + off;
    off += (bytes + 255) & ~(size_t)255;
    return p;
  };
  __bf16* weffT = (__bf16*)carve((size_t)TOT * TOT * 2);
  __bf16* ipwT  = (__bf16*)carve((size_t)SEN * CNN_OUT * 2);
  __bf16* outwT = (__bf16*)carve((size_t)NUM_OUT * OUTD * 2);
  __bf16* feat  = (__bf16*)carve((size_t)BATCH * CNN_OUT * 2);
  float*  E     = (float*) carve((size_t)BATCH * SEN * 4);
  float*  s0    = (float*) carve((size_t)BATCH * TOT * 4);
  float*  s1    = (float*) carve((size_t)BATCH * TOT * 4);
  __bf16* sb0   = (__bf16*)carve((size_t)BATCH * TOT * 2);
  __bf16* sb1   = (__bf16*)carve((size_t)BATCH * TOT * 2);

  build_weff<<<dim3(TOT / 32, TOT / 32), 256, 0, stream>>>(W, lora_A, lora_B, weffT);
  transpose_cvt<<<dim3((SEN + 31) / 32, (CNN_OUT + 31) / 32), 256, 0, stream>>>(
      ip_w, ipwT, CNN_OUT, SEN);
  transpose_cvt<<<dim3((NUM_OUT + 31) / 32, (OUTD + 31) / 32), 256, 0, stream>>>(
      out_w, outwT, OUTD, NUM_OUT);
  conv_feat<<<BATCH, 256, 0, stream>>>(x, cw[0], cw[1], cw[2], cw[3], cw[4],
                                       cw[5], cw[6], cw[7], conv_b, feat);
  // E = feat @ ip_w + ip_b          (1024 x 1024 x 3264, tile-aligned)
  gemm_bf16<false><<<dim3(SEN / TN, BATCH / TM), 256, 0, stream>>>(
      feat, CNN_OUT, ipwT, CNN_OUT, ip_b, nullptr, 0,
      E, SEN, nullptr, 0, BATCH, SEN, CNN_OUT, 0);
  init_state<<<(BATCH * TOT) / 256, 256, 0, stream>>>(E, s0, sb0);
  // 4 recurrent steps: state = relu(state + state @ W_eff)   (tile-aligned)
  float*  sf[2] = {s0, s1};
  __bf16* sb[2] = {sb0, sb1};
  for (int t = 0; t < 4; ++t) {
    int src = t & 1, dst = 1 - src;
    gemm_bf16<false><<<dim3(TOT / TN, BATCH / TM), 256, 0, stream>>>(
        sb[src], TOT, weffT, TOT, nullptr, sf[src], TOT,
        sf[dst], TOT, sb[dst], TOT, BATCH, TOT, TOT, 1);
  }
  // out = state[:, 3072:] @ out_w + out_b   (N=1968 ragged -> bounded)
  gemm_bf16<true><<<dim3((NUM_OUT + TN - 1) / TN, BATCH / TM), 256, 0, stream>>>(
      sb0 + (SEN + 2048), TOT, outwT, OUTD, out_b, nullptr, 0,
      out, NUM_OUT, nullptr, 0, BATCH, NUM_OUT, OUTD, 0);
}
